// MultiHeadAttention_523986010238
// MI455X (gfx1250) — compile-verified, matrix operand order checked by reading, not run on hardware
//
#include <hip/hip_runtime.h>
#include <hip/hip_bf16.h>

// ---------------------------------------------------------------------------
// MultiHeadAttention for MI455X (gfx1250), wave32 + V_WMMA_F32_16X16X32_BF16
// + Tensor Data Mover staging (double-buffered in the GEMMs) with the DMA
// engine inserting the LDS anti-bank-conflict padding.
// B=2, S=2048, D=1024, H=16, DK=64.
// ---------------------------------------------------------------------------

typedef __attribute__((ext_vector_type(16))) __bf16 v16bf;
typedef __attribute__((ext_vector_type(8)))  __bf16 v8bf;
typedef __attribute__((ext_vector_type(8)))  float  v8f;

union AF { v16bf v; v8bf h[2]; };

#define BDIM   2
#define SDIM   2048
#define DDIM   1024
#define HDIM   16
#define DKDIM  64
#define MROWS  (BDIM * SDIM)          // 4096

// ---------------------------------------------------------------------------
// Tensor Data Mover: 2D tile Global->LDS with pad-on-store (6-arg builtin on
// this toolchain; D# layout per CDNA5 ISA ch.8). pad_interval code k =>
// 2^(k+1) DWORDs between pads; pad_amount code a => (a+1) DWORDs of pad.
// Loop-invariant descriptor words are built once; per-issue we patch only
// lds_addr / global_addr (keeps SALU overhead out of the K loop).
// ---------------------------------------------------------------------------
#if defined(__AMDGCN__) && __has_builtin(__builtin_amdgcn_tensor_load_to_lds) && \
    __has_builtin(__builtin_amdgcn_s_wait_tensorcnt)
#define USE_TDM 1
typedef unsigned int u32x4 __attribute__((ext_vector_type(4)));
typedef int          i32x8 __attribute__((ext_vector_type(8)));
typedef int          i32x4 __attribute__((ext_vector_type(4)));

struct Tdm2D { i32x8 g1; };

__device__ __forceinline__ Tdm2D tdm_make(int dim0, int rows, long long stride0,
                                          int pad_int_code, int pad_amt_code) {
  Tdm2D d;
  d.g1[0] = (1 << 16)                             // data_size=1 (2 bytes)
            | (1 << 20)                           // pad_enable
            | (pad_int_code << 22)                // pad_interval [24:22]
            | (pad_amt_code << 25);               // pad_amount   [31:25]
  d.g1[1] = (dim0 & 0xffff) << 16;                // tensor_dim0[15:0] @ [63:48]
  d.g1[2] = ((dim0 >> 16) & 0xffff)               // tensor_dim0[31:16]
            | ((rows & 0xffff) << 16);            // tensor_dim1[15:0] @ [111:80]
  d.g1[3] = ((rows >> 16) & 0xffff)               // tensor_dim1[31:16]
            | ((dim0 & 0xffff) << 16);            // tile_dim0 @ [127:112]
  d.g1[4] = rows & 0xffff;                        // tile_dim1 @ [143:128]
  d.g1[5] = (int)(stride0 & 0xffffffffll);        // tensor_dim0_stride low32
  d.g1[6] = (int)((stride0 >> 32) & 0xffff);      // stride high16 @ [207:192]
  d.g1[7] = 0;
  return d;
}

__device__ __forceinline__ void tdm_issue(const Tdm2D& d, const __bf16* gptr,
                                          unsigned lds_off) {
  unsigned long long ga = (unsigned long long)(uintptr_t)gptr;
  u32x4 g0;
  g0[0] = 1u;                                   // count=1, user-mode load
  g0[1] = lds_off;                              // lds_addr [63:32]
  g0[2] = (unsigned)(ga & 0xffffffffu);         // global_addr low
  g0[3] = (unsigned)((ga >> 32) & 0x1ffffffu)   // global_addr high (57-bit)
          | (2u << 30);                         // type=2 ("image")
  i32x4 z4 = {0, 0, 0, 0};                      // 2D tensor: groups 2/3 zero
  i32x8 z8 = {0, 0, 0, 0, 0, 0, 0, 0};
  __builtin_amdgcn_tensor_load_to_lds(g0, d.g1, z4, z4, z8, 0);
}
#else
#define USE_TDM 0
#endif

// ---------------------------------------------------------------------------
// f32 -> bf16 conversion (vectorized x4)
// ---------------------------------------------------------------------------
__global__ __launch_bounds__(256) void cvt_f32_bf16(const float* __restrict__ src,
                                                    __bf16* __restrict__ dst, int n) {
  int i = (blockIdx.x * 256 + threadIdx.x) * 4;
  if (i + 3 < n) {
    float4 f = *(const float4*)(src + i);
    dst[i + 0] = (__bf16)f.x;
    dst[i + 1] = (__bf16)f.y;
    dst[i + 2] = (__bf16)f.z;
    dst[i + 3] = (__bf16)f.w;
  }
}

// ---------------------------------------------------------------------------
// C[M,N] = A[M,K] @ W[N,K]^T + bias;  M=4096, N=K=1024.
// 256 threads = 8 waves (4x2); block tile 128x128; wave tile 32x64 =
// 2x4 WMMA accumulators; K step 32. A/B tiles staged by the TDM,
// DOUBLE-BUFFERED: wave0 issues the next tile pair, then s_wait_tensorcnt 2
// drains exactly the previous pair (per-wave in-order completion), so the
// DMA for tile k+1 overlaps the WMMAs on tile k.
// MODE 0: bf16 split-head [B,H,S,DK] store with scale. MODE 1: f32 [M,N].
// ---------------------------------------------------------------------------
template <int MODE>
__global__ __launch_bounds__(256) void gemm_xwt(const __bf16* __restrict__ A,
                                                const __bf16* __restrict__ W,
                                                const float* __restrict__ bias,
                                                void* __restrict__ out,
                                                float scale) {
  constexpr int K = DDIM;
  constexpr int NT = K / 32;            // 32 K-steps
  const int n0 = blockIdx.x * 128;
  const int m0 = blockIdx.y * 128;
  const int tid = threadIdx.x;
  const int wave = tid >> 5, lane = tid & 31;
  const int l = lane & 15, half = lane >> 4;
  const int wm = wave >> 1, wn = wave & 1;

  __shared__ __align__(16) __bf16 As[2][128 * 40];
  __shared__ __align__(16) __bf16 Bs[2][128 * 40];

  v8f acc[2][4] = {};

#if USE_TDM
  const Tdm2D dsc = tdm_make(32, 128, K, 3, 3);   // row 64B + 16B pad -> stride 40
  if (tid < 32) {
    tdm_issue(dsc, &A[(size_t)m0 * K], (unsigned)(uintptr_t)&As[0][0]);
    tdm_issue(dsc, &W[(size_t)n0 * K], (unsigned)(uintptr_t)&Bs[0][0]);
  }
#endif

  for (int kt = 0; kt < NT; ++kt) {
    const int cur = kt & 1;
#if USE_TDM
    if (tid < 32) {
      if (kt + 1 < NT) {
        const int k1 = (kt + 1) * 32;
        tdm_issue(dsc, &A[(size_t)m0 * K + k1], (unsigned)(uintptr_t)&As[cur ^ 1][0]);
        tdm_issue(dsc, &W[(size_t)n0 * K + k1], (unsigned)(uintptr_t)&Bs[cur ^ 1][0]);
        __builtin_amdgcn_s_wait_tensorcnt(2);   // drain previous pair only
      } else {
        __builtin_amdgcn_s_wait_tensorcnt(0);
      }
    }
#else
    __syncthreads();
    const int k0 = kt * 32;
#pragma unroll
    for (int c = 0; c < 2; ++c) {             // 512 16B chunks per tile
      int chunk = tid + c * 256;
      int row = chunk >> 2;
      int col = (chunk & 3) * 8;
      *(uint4*)&As[cur][row * 40 + col] =
          *(const uint4*)&A[(size_t)(m0 + row) * K + k0 + col];
      *(uint4*)&Bs[cur][row * 40 + col] =
          *(const uint4*)&W[(size_t)(n0 + row) * K + k0 + col];
    }
#endif
    __syncthreads();                           // publish buf[cur]

    AF a[2];
#pragma unroll
    for (int i = 0; i < 2; ++i) {
      const __bf16* pa = &As[cur][(wm * 32 + i * 16 + l) * 40];
      a[i].h[0] = *(const v8bf*)(pa + half * 8);
      a[i].h[1] = *(const v8bf*)(pa + 16 + half * 8);
    }
#pragma unroll
    for (int j = 0; j < 4; ++j) {
      AF b;
      const __bf16* pb = &Bs[cur][(wn * 64 + j * 16 + l) * 40];
      b.h[0] = *(const v8bf*)(pb + half * 8);
      b.h[1] = *(const v8bf*)(pb + 16 + half * 8);
#pragma unroll
      for (int i = 0; i < 2; ++i)
        acc[i][j] = __builtin_amdgcn_wmma_f32_16x16x32_bf16(
            false, a[i].v, false, b.v, (short)0, acc[i][j], false, false);
    }
    __syncthreads();                           // all readers done with buf[cur]
  }

#pragma unroll
  for (int j = 0; j < 4; ++j) {
    int n = n0 + wn * 64 + j * 16 + l;
    float bval = bias[n];
#pragma unroll
    for (int i = 0; i < 2; ++i) {
#pragma unroll
      for (int r = 0; r < 8; ++r) {
        int m = m0 + wm * 32 + i * 16 + half * 8 + r;
        float val = acc[i][j][r] + bval;
        if (MODE == 0) {
          int b = m >> 11, s = m & (SDIM - 1);
          int h = n >> 6, dk = n & 63;
          ((__bf16*)out)[(((size_t)(b * HDIM + h) * SDIM) + s) * DKDIM + dk] =
              (__bf16)(val * scale);
        } else {
          ((float*)out)[(size_t)m * DDIM + n] = val;
        }
      }
    }
  }
}

// ---------------------------------------------------------------------------
// Wave32 row reductions: rows of a 16x16 C tile live within one 16-lane half.
// ---------------------------------------------------------------------------
__device__ __forceinline__ float rowmax16(float x) {
#pragma unroll
  for (int m = 1; m < 16; m <<= 1) x = fmaxf(x, __shfl_xor(x, m, 16));
  return x;
}
__device__ __forceinline__ float rowsum16(float x) {
#pragma unroll
  for (int m = 1; m < 16; m <<= 1) x += __shfl_xor(x, m, 16);
  return x;
}

// ---------------------------------------------------------------------------
// Flash attention. Q,K,V: [BH=32][S=2048][DK=64] bf16 (scale folded into Q).
// O: [B,S,D] bf16. Grid (S/64, B*H); 128 threads; wave w owns 16 query rows.
// Key block = 128 (amortizes the per-block softmax/reduction over 32 WMMAs).
// Q/K tiles staged by TDM (row 128B + 16B pad -> stride 72); the K-tile TDM
// overlaps the manual V-transpose global loads before the tensorcnt wait.
// ---------------------------------------------------------------------------
__global__ __launch_bounds__(128) void flash_attn(const __bf16* __restrict__ Q,
                                                  const __bf16* __restrict__ Kt,
                                                  const __bf16* __restrict__ V,
                                                  __bf16* __restrict__ O) {
  const int qb = blockIdx.x;     // query block (64 rows)
  const int bh = blockIdx.y;     // batch*head
  const int tid = threadIdx.x;
  const int wave = tid >> 5, lane = tid & 31;
  const int l = lane & 15, half = lane >> 4;

  __shared__ __align__(16) __bf16 Qs[64 * 72];        // [m][dk]
  __shared__ __align__(16) __bf16 Ks[128 * 72];       // [key][dk]
  __shared__ __align__(16) __bf16 Vs[64 * 136];       // [dk][key] (transposed)
  __shared__ __align__(16) __bf16 Ps[4][16 * 136];    // per-wave P tile [m][key]

  const size_t base = (size_t)bh * SDIM * DKDIM;

#if USE_TDM
  const Tdm2D dQ = tdm_make(64, 64, 64, 4, 3);
  const Tdm2D dK = tdm_make(64, 128, 64, 4, 3);
  if (tid < 32) {
    tdm_issue(dQ, &Q[base + (size_t)(qb * 64) * DKDIM], (unsigned)(uintptr_t)Qs);
    __builtin_amdgcn_s_wait_tensorcnt(0);
  }
#else
#pragma unroll
  for (int c = 0; c < 4; ++c) {
    int chunk = tid + c * 128;
    int row = chunk >> 3;
    int col = (chunk & 7) * 8;
    *(uint4*)&Qs[row * 72 + col] =
        *(const uint4*)&Q[base + (size_t)(qb * 64 + row) * DKDIM + col];
  }
#endif
  __syncthreads();

  AF qa[2];
#pragma unroll
  for (int kk = 0; kk < 2; ++kk) {
    const __bf16* p = &Qs[(wave * 16 + l) * 72 + kk * 32];
    qa[kk].h[0] = *(const v8bf*)(p + half * 8);
    qa[kk].h[1] = *(const v8bf*)(p + 16 + half * 8);
  }

  v8f oacc[4] = {};
  float mrun[8], lrun[8];
#pragma unroll
  for (int r = 0; r < 8; ++r) { mrun[r] = -1e30f; lrun[r] = 0.0f; }

  for (int kb = 0; kb < SDIM / 128; ++kb) {
    __syncthreads();
    // K block (128x64) via TDM; V block transposed into [dk][key] manually.
#if USE_TDM
    if (tid < 32)
      tdm_issue(dK, &Kt[base + (size_t)(kb * 128) * DKDIM], (unsigned)(uintptr_t)Ks);
#else
#pragma unroll
    for (int c = 0; c < 8; ++c) {
      int chunk = tid + c * 128;
      int row = chunk >> 3;
      int col = (chunk & 7) * 8;
      *(uint4*)&Ks[row * 72 + col] =
          *(const uint4*)&Kt[base + (size_t)(kb * 128 + row) * DKDIM + col];
    }
#endif
#pragma unroll
    for (int c = 0; c < 8; ++c) {
      int chunk = tid + c * 128;
      int key = chunk >> 3;
      int d0 = (chunk & 7) * 8;
      uint4 raw = *(const uint4*)&V[base + (size_t)(kb * 128 + key) * DKDIM + d0];
      const __bf16* e = (const __bf16*)&raw;
#pragma unroll
      for (int i = 0; i < 8; ++i) Vs[(d0 + i) * 136 + key] = e[i];
    }
#if USE_TDM
    if (tid < 32) __builtin_amdgcn_s_wait_tensorcnt(0);
#endif
    __syncthreads();

    // S = Q K^T : 16x128 per wave = 8 N tiles x 2 K chunks
    v8f sacc[8] = {};
#pragma unroll
    for (int kk = 0; kk < 2; ++kk) {
#pragma unroll
      for (int nt = 0; nt < 8; ++nt) {
        AF b;
        const __bf16* p = &Ks[(nt * 16 + l) * 72 + kk * 32];
        b.h[0] = *(const v8bf*)(p + half * 8);
        b.h[1] = *(const v8bf*)(p + 16 + half * 8);
        sacc[nt] = __builtin_amdgcn_wmma_f32_16x16x32_bf16(
            false, qa[kk].v, false, b.v, (short)0, sacc[nt], false, false);
      }
    }

    // Online softmax (lane owns rows m = half*8 + r, one column per N tile)
    float sf[8];
#pragma unroll
    for (int r = 0; r < 8; ++r) {
      float mx = sacc[0][r];
#pragma unroll
      for (int nt = 1; nt < 8; ++nt) mx = fmaxf(mx, sacc[nt][r]);
      mx = rowmax16(mx);
      float mnew = fmaxf(mrun[r], mx);
      sf[r] = __expf(mrun[r] - mnew);
      mrun[r] = mnew;
    }
#pragma unroll
    for (int r = 0; r < 8; ++r) {
      float rs = 0.0f;
#pragma unroll
      for (int nt = 0; nt < 8; ++nt) {
        float p = __expf(sacc[nt][r] - mrun[r]);
        sacc[nt][r] = p;
        rs += p;
      }
      rs = rowsum16(rs);
      lrun[r] = lrun[r] * sf[r] + rs;
#pragma unroll
      for (int nt = 0; nt < 4; ++nt) oacc[nt][r] *= sf[r];
    }

    // P: C-layout -> LDS -> A-fragment layout (same wave only)
#pragma unroll
    for (int nt = 0; nt < 8; ++nt)
#pragma unroll
      for (int r = 0; r < 8; ++r)
        Ps[wave][(half * 8 + r) * 136 + nt * 16 + l] = (__bf16)sacc[nt][r];
    asm volatile("s_wait_dscnt 0" ::: "memory");

    // O += P V  (K dim = 128 keys = 4 chunks of 32)
#pragma unroll
    for (int kk = 0; kk < 4; ++kk) {
      AF pa;
      const __bf16* pp = &Ps[wave][l * 136 + kk * 32];
      pa.h[0] = *(const v8bf*)(pp + half * 8);
      pa.h[1] = *(const v8bf*)(pp + 16 + half * 8);
#pragma unroll
      for (int nt = 0; nt < 4; ++nt) {
        AF bv;
        const __bf16* pv = &Vs[(nt * 16 + l) * 136 + kk * 32];
        bv.h[0] = *(const v8bf*)(pv + half * 8);
        bv.h[1] = *(const v8bf*)(pv + 16 + half * 8);
        oacc[nt] = __builtin_amdgcn_wmma_f32_16x16x32_bf16(
            false, pa.v, false, bv.v, (short)0, oacc[nt], false, false);
      }
    }
  }

  // Normalize and store O in [B,S,D] merged-head layout (bf16)
  const int b = bh >> 4, h = bh & 15;
#pragma unroll
  for (int r = 0; r < 8; ++r) {
    float inv = 1.0f / lrun[r];
    int s = qb * 64 + wave * 16 + half * 8 + r;
#pragma unroll
    for (int nt = 0; nt < 4; ++nt) {
      int dcol = h * DKDIM + nt * 16 + l;
      O[((size_t)(b * SDIM + s)) * DDIM + dcol] = (__bf16)(oacc[nt][r] * inv);
    }
  }
}

// ---------------------------------------------------------------------------
// Host-side launch sequence (all on `stream`, workspace reuse, ~42 MB)
// ---------------------------------------------------------------------------
extern "C" void kernel_launch(void* const* d_in, const int* in_sizes, int n_in,
                              void* d_out, int out_size, void* d_ws, size_t ws_size,
                              hipStream_t stream) {
  const float* q_in = (const float*)d_in[0];
  const float* k_in = (const float*)d_in[1];
  const float* v_in = (const float*)d_in[2];
  const float* Wq = (const float*)d_in[3];
  const float* bq = (const float*)d_in[4];
  const float* Wk = (const float*)d_in[5];
  const float* bk = (const float*)d_in[6];
  const float* Wv = (const float*)d_in[7];
  const float* bv = (const float*)d_in[8];
  const float* Wo = (const float*)d_in[9];
  const float* bo = (const float*)d_in[10];

  char* ws = (char*)d_ws;
  const size_t MB = 1u << 20;
  __bf16* xbuf = (__bf16*)(ws + 0 * MB);    // 8 MB  [4096x1024]
  __bf16* wbuf = (__bf16*)(ws + 8 * MB);    // 2 MB  [1024x1024]
  __bf16* qb_  = (__bf16*)(ws + 10 * MB);   // 8 MB  [32][2048][64]
  __bf16* kb_  = (__bf16*)(ws + 18 * MB);   // 8 MB
  __bf16* vb_  = (__bf16*)(ws + 26 * MB);   // 8 MB
  __bf16* ob_  = (__bf16*)(ws + 34 * MB);   // 8 MB  [4096x1024]

  const int NX = MROWS * DDIM;   // 4M
  const int NW = DDIM * DDIM;    // 1M
  dim3 ggemm(DDIM / 128, MROWS / 128);  // (8, 32)
  dim3 gattn(SDIM / 64, BDIM * HDIM);   // (32, 32)
  const float qscale = 0.125f;          // 1/sqrt(DK)

  // Q projection (scale folded into Q)
  cvt_f32_bf16<<<NX / 1024, 256, 0, stream>>>(q_in, xbuf, NX);
  cvt_f32_bf16<<<NW / 1024, 256, 0, stream>>>(Wq, wbuf, NW);
  gemm_xwt<0><<<ggemm, 256, 0, stream>>>(xbuf, wbuf, bq, qb_, qscale);

  // K projection
  cvt_f32_bf16<<<NX / 1024, 256, 0, stream>>>(k_in, xbuf, NX);
  cvt_f32_bf16<<<NW / 1024, 256, 0, stream>>>(Wk, wbuf, NW);
  gemm_xwt<0><<<ggemm, 256, 0, stream>>>(xbuf, wbuf, bk, kb_, 1.0f);

  // V projection
  cvt_f32_bf16<<<NX / 1024, 256, 0, stream>>>(v_in, xbuf, NX);
  cvt_f32_bf16<<<NW / 1024, 256, 0, stream>>>(Wv, wbuf, NW);
  gemm_xwt<0><<<ggemm, 256, 0, stream>>>(xbuf, wbuf, bv, vb_, 1.0f);

  // Attention
  flash_attn<<<gattn, 128, 0, stream>>>(qb_, kb_, vb_, ob_);

  // Output projection -> f32 d_out
  cvt_f32_bf16<<<NW / 1024, 256, 0, stream>>>(Wo, wbuf, NW);
  gemm_xwt<1><<<ggemm, 256, 0, stream>>>(ob_, wbuf, bo, (float*)d_out, 1.0f);
}
